// Goal_Conditioned_Policies_87978110092002
// MI455X (gfx1250) — compile-verified, matrix operand order checked by reading, not run on hardware
//
#include <hip/hip_runtime.h>

// ---------------------------------------------------------------------------
// CDNA5 (gfx1250, wave32) fused implementation of the conv-encoder + MLP head.
//   Kernel 0 (prep): weights -> f16, transposed [N][K], zero-padded.
//   Kernel 1: per-sample conv1 (VALU) + conv2 (WMMA, 9x K=32) + mean pool.
//             h1 kept halo-padded in LDS so all WMMA A-fragment loads are
//             branch-free ds_load_b128 pairs.
//   Kernel 2: per-16-row fused MLP chain, all GEMMs on v_wmma_f32_16x16x32_f16.
// ---------------------------------------------------------------------------

typedef __attribute__((ext_vector_type(16))) _Float16 v16h;
typedef __attribute__((ext_vector_type(8)))  float    v8f;
typedef __attribute__((ext_vector_type(4)))  float    v4f;

// Load a 16x32 f16 A-fragment (or 32x16 B-fragment, they mirror) from a
// row-major [16+][ld] f16 matrix starting at column k0.
// Per ISA 7.12.2: lane<16 holds K = k0+{0..7} and k0+{16..23};
//                 lane>=16 holds K = k0+{8..15} and k0+{24..31}.
static __device__ __forceinline__ v16h load_frag(const _Float16* __restrict__ base,
                                                 int ld, int k0) {
  const int lane = threadIdx.x & 31;
  const int row  = lane & 15;
  const int hi   = lane >> 4;
  const _Float16* p = base + (size_t)row * ld + k0 + hi * 8;
  v16h f;
  ((v4f*)&f)[0] = *(const v4f*)p;          // K chunk 0 (16 bytes = 8 halves)
  ((v4f*)&f)[1] = *(const v4f*)(p + 16);   // K chunk 1
  return f;
}

// Broadcast bias into a C-fragment: element (M,N) gets bias[N]; N = n0+(lane&15).
static __device__ __forceinline__ v8f bias_acc(const float* __restrict__ b,
                                               int n0, int nvalid) {
  const int n = n0 + (threadIdx.x & 15);
  const float v = (n < nvalid) ? b[n] : 0.f;
  v8f c;
#pragma unroll
  for (int i = 0; i < 8; ++i) c[i] = v;
  return c;
}

// Store D-fragment (relu) as f16 into row-major [16][ld] at column block n0.
// D layout: VGPR r -> row r (lanes 0-15) / row r+8 (lanes 16-31), col = lane&15.
static __device__ __forceinline__ void store_relu_h(_Float16* __restrict__ dst,
                                                    int ld, int n0, v8f c) {
  const int lane = threadIdx.x & 31;
  const int col  = n0 + (lane & 15);
  const int rb   = (lane >> 4) * 8;
#pragma unroll
  for (int r = 0; r < 8; ++r) {
    float v = c[r];
    dst[(rb + r) * ld + col] = (_Float16)(v > 0.f ? v : 0.f);
  }
}

static __device__ __forceinline__ void store_relu_f(float* __restrict__ dst,
                                                    int ld, int n0, v8f c) {
  const int lane = threadIdx.x & 31;
  const int col  = n0 + (lane & 15);
  const int rb   = (lane >> 4) * 8;
#pragma unroll
  for (int r = 0; r < 8; ++r) {
    float v = c[r];
    dst[(rb + r) * ld + col] = (v > 0.f ? v : 0.f);
  }
}

static __device__ __forceinline__ void store_f(float* __restrict__ dst,
                                               int ld, int n0, v8f c) {
  const int lane = threadIdx.x & 31;
  const int col  = n0 + (lane & 15);
  const int rb   = (lane >> 4) * 8;
#pragma unroll
  for (int r = 0; r < 8; ++r) dst[(rb + r) * ld + col] = c[r];
}

#define WMMA_F16(a, b, c) \
  __builtin_amdgcn_wmma_f32_16x16x32_f16(false, (a), false, (b), (short)0, (c), false, false)

// ---------------------------------------------------------------------------
// Kernel 0a: transpose + f16-convert + zero-pad FC weight.
// src: f32 [K][N] row-major.  dst: f16 [Np][Kp], dst[n][k] = src[k][n] (0 if padded).
// ---------------------------------------------------------------------------
__global__ void prep_wt_kernel(const float* __restrict__ src, _Float16* __restrict__ dst,
                               int K, int N, int Kp, int Np) {
  int idx = blockIdx.x * blockDim.x + threadIdx.x;
  if (idx >= Np * Kp) return;
  int n = idx / Kp, k = idx - n * Kp;
  float v = (n < N && k < K) ? src[k * N + n] : 0.f;
  dst[idx] = (_Float16)v;
}

// Kernel 0b: conv2 weights [64][32][3][3] (OIHW) -> f16 [tap][co][ci].
__global__ void prep_w2_kernel(const float* __restrict__ w2, _Float16* __restrict__ dst) {
  int idx = blockIdx.x * blockDim.x + threadIdx.x;
  if (idx >= 9 * 64 * 32) return;
  int tap = idx / (64 * 32);
  int co  = (idx / 32) % 64;
  int ci  = idx & 31;
  dst[idx] = (_Float16)w2[(co * 32 + ci) * 9 + tap];
}

// ---------------------------------------------------------------------------
// Kernel 1: one workgroup per sample.
//   LDS: xs [3][64][64] f16, h1p [34][34][32] f16 (halo-padded, zero border),
//        w2s [9][64][32] f16, w1s/b1s/pool f32.
// ---------------------------------------------------------------------------
#define H1LD 34                       // padded spatial extent of h1
#define H1_ELEMS (H1LD * H1LD * 32)   // 36992 halves

__global__ __launch_bounds__(256) void conv_encoder_kernel(
    const float* __restrict__ x,      // [B,3,64,64]
    const float* __restrict__ w1,     // [32,3,3,3]
    const float* __restrict__ b1,     // [32]
    const _Float16* __restrict__ w2t, // [9][64][32] f16 (ws)
    const float* __restrict__ b2,     // [64]
    _Float16* __restrict__ pooled) {  // [B][64] f16 (ws)
  extern __shared__ char smem[];
  _Float16* xs   = (_Float16*)smem;            // 12288 halves
  _Float16* h1p  = xs + 3 * 64 * 64;           // 36992 halves (halo padded)
  _Float16* w2s  = h1p + H1_ELEMS;             // 18432 halves
  float*    w1s  = (float*)(w2s + 9 * 64 * 32); // [32][27]
  float*    b1s  = w1s + 32 * 27;              // [32]
  float*    pool = b1s + 32;                   // [64]

  const int b   = blockIdx.x;
  const int tid = threadIdx.x;
  const float* xb = x + (size_t)b * 3 * 64 * 64;

  // ---- staging (vectorized): x f32->f16, w2 copy, h1 halo zero-fill ----
  for (int i4 = tid; i4 < 3 * 64 * 64 / 4; i4 += 256) {     // 16B global loads
    v4f v = ((const v4f*)xb)[i4];
    _Float16* d = xs + i4 * 4;
    d[0] = (_Float16)v[0]; d[1] = (_Float16)v[1];
    d[2] = (_Float16)v[2]; d[3] = (_Float16)v[3];
  }
  for (int i4 = tid; i4 < 9 * 64 * 32 / 8; i4 += 256)       // 16B LDS stores
    ((v4f*)w2s)[i4] = ((const v4f*)w2t)[i4];
  {
    v4f z; z[0] = 0.f; z[1] = 0.f; z[2] = 0.f; z[3] = 0.f;
    for (int i4 = tid; i4 < H1_ELEMS / 8; i4 += 256)        // zero incl. halo
      ((v4f*)h1p)[i4] = z;
  }
  for (int i = tid; i < 32 * 27; i += 256) w1s[i] = w1[i];
  if (tid < 32) b1s[tid] = b1[tid];
  if (tid < 64) pool[tid] = 0.f;
  __syncthreads();

  // ---- conv1 direct (K=27), relu, store channel-last f16 at halo offset ----
  for (int p = tid; p < 32 * 32; p += 256) {
    const int oy = p >> 5, ox = p & 31;
    float in[27];
#pragma unroll
    for (int ci = 0; ci < 3; ++ci)
#pragma unroll
      for (int ky = 0; ky < 3; ++ky)
#pragma unroll
        for (int kx = 0; kx < 3; ++kx) {
          const int iy = 2 * oy - 1 + ky, ix = 2 * ox - 1 + kx;
          in[ci * 9 + ky * 3 + kx] =
              (iy >= 0 && iy < 64 && ix >= 0 && ix < 64)
                  ? (float)xs[(ci * 64 + iy) * 64 + ix] : 0.f;
        }
    _Float16* h1w = h1p + (((oy + 1) * H1LD) + (ox + 1)) * 32;
    for (int co = 0; co < 32; ++co) {
      float s = b1s[co];
      const float* wr = w1s + co * 27;
#pragma unroll
      for (int k = 0; k < 27; ++k) s = fmaf(in[k], wr[k], s);
      h1w[co] = (_Float16)(s > 0.f ? s : 0.f);
    }
  }
  __syncthreads();

  // ---- conv2 as 9 accumulated K=32 WMMA GEMMs (branch-free frag loads) ----
  const int wave = tid >> 5;
  const int lane = tid & 31;
  const int lm   = lane & 15;
  const int hi   = lane >> 4;

  v8f acc[2][4];
#pragma unroll
  for (int mt = 0; mt < 2; ++mt)
#pragma unroll
    for (int nt = 0; nt < 4; ++nt) {
      const float bv = b2[nt * 16 + lm];
#pragma unroll
      for (int i = 0; i < 8; ++i) acc[mt][nt][i] = bv;
    }

#pragma unroll
  for (int tap = 0; tap < 9; ++tap) {
    const int ky = tap / 3, kx = tap % 3;
    v16h afrag[2];
#pragma unroll
    for (int mt = 0; mt < 2; ++mt) {
      const int p  = (wave * 2 + mt) * 16 + lm;   // output pixel 0..255
      const int oy = p >> 4, ox = p & 15;
      // padded coords: (2*oy-1+ky)+1, (2*ox-1+kx)+1  -> always in [0, 34)
      const int iy = 2 * oy + ky, ix = 2 * ox + kx;
      const _Float16* base = h1p + ((iy * H1LD + ix) * 32) + hi * 8;
      v16h a;
      ((v4f*)&a)[0] = *(const v4f*)base;
      ((v4f*)&a)[1] = *(const v4f*)(base + 16);
      afrag[mt] = a;
    }
#pragma unroll
    for (int nt = 0; nt < 4; ++nt) {
      const _Float16* wb = w2s + ((tap * 64) + nt * 16 + lm) * 32 + hi * 8;
      v16h bfrag;
      ((v4f*)&bfrag)[0] = *(const v4f*)wb;
      ((v4f*)&bfrag)[1] = *(const v4f*)(wb + 16);
#pragma unroll
      for (int mt = 0; mt < 2; ++mt)
        acc[mt][nt] = WMMA_F16(afrag[mt], bfrag, acc[mt][nt]);
    }
  }

  // ---- relu + mean pool (per-lane partial sums over M, ds_add_f32) ----
#pragma unroll
  for (int mt = 0; mt < 2; ++mt)
#pragma unroll
    for (int nt = 0; nt < 4; ++nt) {
      float s = 0.f;
#pragma unroll
      for (int r = 0; r < 8; ++r) {
        float v = acc[mt][nt][r];
        s += (v > 0.f ? v : 0.f);
      }
      atomicAdd(&pool[nt * 16 + lm], s);
    }
  __syncthreads();
  if (tid < 64) pooled[(size_t)b * 64 + tid] = (_Float16)(pool[tid] * (1.f / 256.f));
}

// ---------------------------------------------------------------------------
// Kernel 2: one wave (32 threads) per 16 batch rows; full fused MLP chain.
// ---------------------------------------------------------------------------
__global__ __launch_bounds__(32) void mlp_chain_kernel(
    const _Float16* __restrict__ pooled,  // [4096][64]
    const float* __restrict__ C,          // [4096][50][50]
    const _Float16* __restrict__ wt_enc,  // [128][64]
    const float* __restrict__ enc_b,
    const _Float16* __restrict__ wt_f1,   // [64][128]  (rows>=50 zero)
    const float* __restrict__ f1_b,
    const _Float16* __restrict__ wt_f2,   // [128][64]  (cols>=50 zero)
    const float* __restrict__ f2_b,
    const _Float16* __restrict__ wt_f3,   // [64][256]
    const float* __restrict__ f3_b,
    const _Float16* __restrict__ wt_f4,   // [64][64]
    const float* __restrict__ f4_b,
    const _Float16* __restrict__ wt_f5,   // [64][64]
    const float* __restrict__ f5_b,
    const _Float16* __restrict__ wt_out,  // [64][64]  (rows>=50 zero)
    const float* __restrict__ out_b,
    float* __restrict__ out) {            // [4096][50]
  __shared__ _Float16 sA[16][256];  // x1 (cols 0..127) | x3 (cols 128..255)
  __shared__ _Float16 sB[16][64];   // e, then x5/x6/x7
  __shared__ float    sE[16][64];   // x2 f32, then logits

  const int r0   = blockIdx.x * 16;
  const int lane = threadIdx.x;

  // ---- enc: x1 = relu(pooled @ enc_w + b), M=16 N=128 K=64 ----
  {
    const _Float16* A = pooled + (size_t)r0 * 64;
    v16h a0 = load_frag(A, 64, 0);
    v16h a1 = load_frag(A, 64, 32);
#pragma unroll
    for (int nt = 0; nt < 8; ++nt) {
      v8f c = bias_acc(enc_b, nt * 16, 128);
      c = WMMA_F16(a0, load_frag(wt_enc + nt * 16 * 64, 64, 0),  c);
      c = WMMA_F16(a1, load_frag(wt_enc + nt * 16 * 64, 64, 32), c);
      store_relu_h(&sA[0][0], 256, nt * 16, c);
    }
  }
  __syncthreads();

  // ---- f1: x2 = relu(x1 @ f1_w + b), N=50 (pad 64) K=128 -> sE f32 ----
  {
    v16h a[4];
#pragma unroll
    for (int kt = 0; kt < 4; ++kt) a[kt] = load_frag(&sA[0][0], 256, kt * 32);
#pragma unroll
    for (int nt = 0; nt < 4; ++nt) {
      v8f c = bias_acc(f1_b, nt * 16, 50);
#pragma unroll
      for (int kt = 0; kt < 4; ++kt)
        c = WMMA_F16(a[kt], load_frag(wt_f1 + nt * 16 * 128, 128, kt * 32), c);
      store_relu_f(&sE[0][0], 64, nt * 16, c);
    }
  }
  __syncthreads();

  // ---- batched matvec: e[b] = C[b] @ x2[b]   (bandwidth bound: C = 41 MB) ----
  for (int i = lane; i < 16 * 64; i += 32) (&sB[0][0])[i] = (_Float16)0.f;
  for (int o = lane; o < 16 * 50; o += 32) {
    const int row = o / 50, i = o - row * 50;
    const float* crow = C + ((size_t)(r0 + row) * 50 + i) * 50;
    float s = 0.f;
    for (int j = 0; j < 50; ++j) s = fmaf(crow[j], sE[row][j], s);
    sB[row][i] = (_Float16)s;
  }
  __syncthreads();

  // ---- f2: x3 = relu(e @ f2_w + b), N=128 K=50(pad 64) -> sA cols 128..255 ----
  {
    v16h a0 = load_frag(&sB[0][0], 64, 0);
    v16h a1 = load_frag(&sB[0][0], 64, 32);
#pragma unroll
    for (int nt = 0; nt < 8; ++nt) {
      v8f c = bias_acc(f2_b, nt * 16, 128);
      c = WMMA_F16(a0, load_frag(wt_f2 + nt * 16 * 64, 64, 0),  c);
      c = WMMA_F16(a1, load_frag(wt_f2 + nt * 16 * 64, 64, 32), c);
      store_relu_h(&sA[0][0], 256, 128 + nt * 16, c);
    }
  }
  __syncthreads();

  // ---- f3: x5 = relu(concat(x1,x3) @ f3_w + b), N=64 K=256 -> sB ----
  {
    v16h a[8];
#pragma unroll
    for (int kt = 0; kt < 8; ++kt) a[kt] = load_frag(&sA[0][0], 256, kt * 32);
#pragma unroll
    for (int nt = 0; nt < 4; ++nt) {
      v8f c = bias_acc(f3_b, nt * 16, 64);
#pragma unroll
      for (int kt = 0; kt < 8; ++kt)
        c = WMMA_F16(a[kt], load_frag(wt_f3 + nt * 16 * 256, 256, kt * 32), c);
      store_relu_h(&sB[0][0], 64, nt * 16, c);
    }
  }
  __syncthreads();

  // ---- f4 then f5: 64x64, in-place through sB (compute->barrier->store) ----
#pragma unroll
  for (int layer = 0; layer < 2; ++layer) {
    const _Float16* wt = layer ? wt_f5 : wt_f4;
    const float*    bb = layer ? f5_b  : f4_b;
    v16h a0 = load_frag(&sB[0][0], 64, 0);
    v16h a1 = load_frag(&sB[0][0], 64, 32);
    v8f cc[4];
#pragma unroll
    for (int nt = 0; nt < 4; ++nt) {
      v8f c = bias_acc(bb, nt * 16, 64);
      c = WMMA_F16(a0, load_frag(wt + nt * 16 * 64, 64, 0),  c);
      c = WMMA_F16(a1, load_frag(wt + nt * 16 * 64, 64, 32), c);
      cc[nt] = c;
    }
    __syncthreads();
#pragma unroll
    for (int nt = 0; nt < 4; ++nt) store_relu_h(&sB[0][0], 64, nt * 16, cc[nt]);
    __syncthreads();
  }

  // ---- out: logits = x7 @ out_w + b, N=50 (pad 64) -> sE f32 ----
  {
    v16h a0 = load_frag(&sB[0][0], 64, 0);
    v16h a1 = load_frag(&sB[0][0], 64, 32);
#pragma unroll
    for (int nt = 0; nt < 4; ++nt) {
      v8f c = bias_acc(out_b, nt * 16, 50);
      c = WMMA_F16(a0, load_frag(wt_out + nt * 16 * 64, 64, 0),  c);
      c = WMMA_F16(a1, load_frag(wt_out + nt * 16 * 64, 64, 32), c);
      store_f(&sE[0][0], 64, nt * 16, c);
    }
  }
  __syncthreads();

  // ---- softmax over 50, one row per lane (lanes 0..15) ----
  if (lane < 16) {
    const int row = lane;
    float m = -3.402823466e+38f;
    for (int i = 0; i < 50; ++i) m = fmaxf(m, sE[row][i]);
    float s = 0.f;
    for (int i = 0; i < 50; ++i) {
      const float e = __expf(sE[row][i] - m);
      sE[row][i] = e;
      s += e;
    }
    const float inv = 1.f / s;
    float* o = out + (size_t)(r0 + row) * 50;
    for (int i = 0; i < 50; ++i) o[i] = sE[row][i] * inv;
  }
}

// ---------------------------------------------------------------------------
// Host launcher.
// ---------------------------------------------------------------------------
extern "C" void kernel_launch(void* const* d_in, const int* in_sizes, int n_in,
                              void* d_out, int out_size, void* d_ws, size_t ws_size,
                              hipStream_t stream) {
  (void)in_sizes; (void)n_in; (void)out_size; (void)ws_size;
  const float* x       = (const float*)d_in[0];
  const float* C       = (const float*)d_in[1];
  const float* conv1_w = (const float*)d_in[2];
  const float* conv1_b = (const float*)d_in[3];
  const float* conv2_w = (const float*)d_in[4];
  const float* conv2_b = (const float*)d_in[5];
  const float* enc_w   = (const float*)d_in[6];
  const float* enc_b   = (const float*)d_in[7];
  const float* f1_w    = (const float*)d_in[8];
  const float* f1_b    = (const float*)d_in[9];
  const float* f2_w    = (const float*)d_in[10];
  const float* f2_b    = (const float*)d_in[11];
  const float* f3_w    = (const float*)d_in[12];
  const float* f3_b    = (const float*)d_in[13];
  const float* f4_w    = (const float*)d_in[14];
  const float* f4_b    = (const float*)d_in[15];
  const float* f5_w    = (const float*)d_in[16];
  const float* f5_b    = (const float*)d_in[17];
  const float* out_w   = (const float*)d_in[18];
  const float* out_b   = (const float*)d_in[19];
  float* out = (float*)d_out;

  // Workspace layout (f16 units).
  _Float16* wsh     = (_Float16*)d_ws;
  _Float16* wt_enc  = wsh;            // [128][64]   8192
  _Float16* wt_f1   = wsh + 8192;     // [64][128]   8192
  _Float16* wt_f2   = wsh + 16384;    // [128][64]   8192
  _Float16* wt_f3   = wsh + 24576;    // [64][256]  16384
  _Float16* wt_f4   = wsh + 40960;    // [64][64]    4096
  _Float16* wt_f5   = wsh + 45056;    // [64][64]    4096
  _Float16* wt_out  = wsh + 49152;    // [64][64]    4096
  _Float16* w2t     = wsh + 53248;    // [9][64][32] 18432
  _Float16* pooled  = wsh + 71680;    // [4096][64] 262144

  // ---- weight prep ----
  prep_wt_kernel<<<(128 * 64 + 255) / 256, 256, 0, stream>>>(enc_w, wt_enc,  64, 128,  64, 128);
  prep_wt_kernel<<<( 64 * 128 + 255) / 256, 256, 0, stream>>>(f1_w,  wt_f1, 128,  50, 128,  64);
  prep_wt_kernel<<<(128 * 64 + 255) / 256, 256, 0, stream>>>(f2_w,  wt_f2,  50, 128,  64, 128);
  prep_wt_kernel<<<( 64 * 256 + 255) / 256, 256, 0, stream>>>(f3_w,  wt_f3, 256,  64, 256,  64);
  prep_wt_kernel<<<( 64 * 64 + 255) / 256, 256, 0, stream>>>(f4_w,  wt_f4,  64,  64,  64,  64);
  prep_wt_kernel<<<( 64 * 64 + 255) / 256, 256, 0, stream>>>(f5_w,  wt_f5,  64,  64,  64,  64);
  prep_wt_kernel<<<( 64 * 64 + 255) / 256, 256, 0, stream>>>(out_w, wt_out,  64,  50,  64,  64);
  prep_w2_kernel<<<(9 * 64 * 32 + 255) / 256, 256, 0, stream>>>(conv2_w, w2t);

  // ---- conv encoder: one WGP per sample; dynamic LDS:
  // (12288 + 36992 + 18432)*2 + (864 + 32 + 64)*4 = 139264 bytes (< 320 KB WGP LDS)
  const size_t conv_lds =
      (size_t)(3 * 64 * 64 + H1_ELEMS + 9 * 64 * 32) * 2 + (size_t)(32 * 27 + 32 + 64) * 4;
  conv_encoder_kernel<<<4096, 256, conv_lds, stream>>>(x, conv1_w, conv1_b, w2t, conv2_b, pooled);

  // ---- fused MLP chain: 256 waves, 16 batch rows each ----
  mlp_chain_kernel<<<4096 / 16, 32, 0, stream>>>(
      pooled, C, wt_enc, enc_b, wt_f1, f1_b, wt_f2, f2_b, wt_f3, f3_b,
      wt_f4, f4_b, wt_f5, f5_b, wt_out, out_b, out);
}
